// AGRAN_anchor_59854664237408
// MI455X (gfx1250) — matrix operand order repaired and checked by reading, not compiled, not run on hardware
//
#include <hip/hip_runtime.h>
#include <hip/hip_bf16.h>
#include <stdint.h>

// ---------------------------------------------------------------------------
// Model constants (match reference)
// ---------------------------------------------------------------------------
#define B_    4
#define L_    200
#define H_    256
#define NH_   4
#define HS_   64
#define NB_   2
#define C_    257            // relative table rows
#define CPAD_ 272            // padded C for GEMM ldc
#define BL_   (B_ * L_)      // 800
#define BLH_  (B_ * L_ * H_) // 204800
#define NEGF  (-4294967295.0f)
#define EPS_  1e-8f

typedef __attribute__((ext_vector_type(16))) __bf16 v16bf;
typedef __attribute__((ext_vector_type(8)))  float  v8f;
typedef __attribute__((ext_vector_type(4)))  unsigned int u32x4;
typedef __attribute__((ext_vector_type(8)))  int    i32x8;
typedef __attribute__((ext_vector_type(4)))  int    i32x4;

#ifndef USE_TDM
#define USE_TDM 1
#endif
#if defined(__clang_major__) && (__clang_major__ >= 23)
#define TDM_SIX_ARGS 1
#else
#define TDM_SIX_ARGS 0
#endif

// ---------------------------------------------------------------------------
// TDM: 2D tile load (fp32) global -> LDS.  Descriptor per CDNA5 ISA §8.3/8.4.
// tensor_dim = remaining extent from the tile base, so HW zero-fills OOB.
// Issued by a single wave (EXEC is ignored by tensor ops).
// ---------------------------------------------------------------------------
#if USE_TDM
__device__ __forceinline__ void tdm_load_tile_2d(const float* gbase,
                                                 unsigned lds_byte_off,
                                                 unsigned tile_w, unsigned tile_h,
                                                 long rem_w, long rem_h,
                                                 unsigned row_stride_elems) {
  unsigned long long ga = (unsigned long long)(uintptr_t)gbase;
  unsigned rw = (rem_w < 0) ? 0u : (unsigned)rem_w;
  unsigned rh = (rem_h < 0) ? 0u : (unsigned)rem_h;
  u32x4 g0;
  g0[0] = 1u;                                        // count=1, user mode
  g0[1] = lds_byte_off;                              // lds_addr
  g0[2] = (unsigned)(ga & 0xffffffffu);              // global_addr[31:0]
  g0[3] = (unsigned)((ga >> 32) & 0x01ffffffu)       // global_addr[56:32]
        | (2u << 30);                                // type = 2 ("image")
  unsigned long long st = (unsigned long long)row_stride_elems;
  i32x8 g1;
  g1[0] = (int)(2u << 16);                           // data_size=2 -> 4 bytes
  g1[1] = (int)((rw & 0xffffu) << 16);               // tensor_dim0[15:0]
  g1[2] = (int)((rw >> 16) | ((rh & 0xffffu) << 16));// dim0[31:16] | dim1[15:0]
  g1[3] = (int)((rh >> 16) | (tile_w << 16));        // dim1[31:16] | tile_dim0
  g1[4] = (int)(tile_h & 0xffffu);                   // tile_dim1 (tile_dim2=0)
  g1[5] = (int)(st & 0xffffffffull);                 // dim0_stride[31:0]
  g1[6] = (int)((st >> 32) & 0xffffull);             // dim0_stride[47:32]
  g1[7] = 0;
  i32x4 z4 = {0, 0, 0, 0};
#if TDM_SIX_ARGS
  i32x8 z8 = {0, 0, 0, 0, 0, 0, 0, 0};
  __builtin_amdgcn_tensor_load_to_lds(g0, g1, z4, z4, z8, 0);
#else
  __builtin_amdgcn_tensor_load_to_lds(g0, g1, z4, z4, 0);
#endif
}
#endif

// ---------------------------------------------------------------------------
// Batched WMMA GEMM:  C = alpha * A x op(B) (+ bias) (+ C) (relu?)
//   A: fp32, row-major M x K, leading dim lda, batch bases A + h*sAh + b*sAb
//   B: fp32; transB=1 -> B is N x K (C = A*B^T), transB=0 -> B is K x N
//   batch z in [0, gridDim.z): h = z / nb, b = z % nb
// Block (4 waves) computes a 32(M) x 64(N) tile; K advances by 64/iteration.
// Per wave: two stacked 16x16 C tiles; per iteration 2 K-substeps x 2 M-sub-
// tiles = 4 v_wmma_f32_16x16x32_bf16, with the B fragment reused across the
// M-subtiles.  fp32 panels are staged in LDS by the Tensor Data Mover with
// double buffering: the next panel's DMA overlaps this iteration's WMMAs.
// ---------------------------------------------------------------------------
__global__ __launch_bounds__(128) void k_gemm(
    const float* __restrict__ A, int lda, long sAh, long sAb,
    const float* __restrict__ Bm, int ldb, long sBh, long sBb, int transB,
    const float* __restrict__ bias,
    float* __restrict__ C, int ldc, long sCh, long sCb,
    int M, int N, int K, float alpha, int accum, int relu, int nb) {
  __shared__ float ldsA[2][32 * 64];   // [buf][m][k]          16 KB
  __shared__ float ldsB[2][64 * 64];   // [buf][n][k] or [k][n] 32 KB

  const int tid  = threadIdx.x;
  const int wave = tid >> 5;
  const int lane = tid & 31;
  const int z = blockIdx.z;
  const int h = z / nb, b = z % nb;
  const float* Ab = A  + (long)h * sAh + (long)b * sAb;
  const float* Bb = Bm + (long)h * sBh + (long)b * sBb;
  float*       Cb = C  + (long)h * sCh + (long)b * sCb;
  const int m0 = blockIdx.x * 32;
  const int n0 = blockIdx.y * 64;

  const int r  = lane & 15;
  const int hi = lane >> 4;
  v8f acc0 = {0.f, 0.f, 0.f, 0.f, 0.f, 0.f, 0.f, 0.f};
  v8f acc1 = {0.f, 0.f, 0.f, 0.f, 0.f, 0.f, 0.f, 0.f};

#if USE_TDM
  // Prologue: stage K-panel 0 into buffer 0.
  if (wave == 0) {
    tdm_load_tile_2d(Ab + (long)m0 * lda, (unsigned)(uintptr_t)&ldsA[0][0],
                     64, 32, (long)K, (long)M - m0, (unsigned)lda);
    if (transB)
      tdm_load_tile_2d(Bb + (long)n0 * ldb, (unsigned)(uintptr_t)&ldsB[0][0],
                       64, 64, (long)K, (long)N - n0, (unsigned)ldb);
    else
      tdm_load_tile_2d(Bb + n0, (unsigned)(uintptr_t)&ldsB[0][0],
                       64, 64, (long)N - n0, (long)K, (unsigned)ldb);
  }
#endif

  int buf = 0;
  for (int k0 = 0; k0 < K; k0 += 64) {
#if USE_TDM
    if (wave == 0) __builtin_amdgcn_s_wait_tensorcnt(0);
    __syncthreads();                 // buffer `buf` ready for everyone
    if (wave == 0 && (k0 + 64) < K) {
      int kn = k0 + 64, nx = buf ^ 1;
      tdm_load_tile_2d(Ab + (long)m0 * lda + kn,
                       (unsigned)(uintptr_t)&ldsA[nx][0], 64, 32,
                       (long)K - kn, (long)M - m0, (unsigned)lda);
      if (transB)
        tdm_load_tile_2d(Bb + (long)n0 * ldb + kn,
                         (unsigned)(uintptr_t)&ldsB[nx][0], 64, 64,
                         (long)K - kn, (long)N - n0, (unsigned)ldb);
      else
        tdm_load_tile_2d(Bb + (long)kn * ldb + n0,
                         (unsigned)(uintptr_t)&ldsB[nx][0], 64, 64,
                         (long)N - n0, (long)K - kn, (unsigned)ldb);
    }
#else
    __syncthreads();
    for (int i = tid; i < 32 * 64; i += 128) {
      int rr = i >> 6, cc = i & 63;
      int gm = m0 + rr, gk = k0 + cc;
      ldsA[buf][i] = (gm < M && gk < K) ? Ab[(long)gm * lda + gk] : 0.f;
    }
    for (int i = tid; i < 64 * 64; i += 128) {
      int rr = i >> 6, cc = i & 63;
      float v = 0.f;
      if (transB) {                  // [n][k]
        int gn = n0 + rr, gk = k0 + cc;
        if (gn < N && gk < K) v = Bb[(long)gn * ldb + gk];
      } else {                       // [k][n]
        int gk = k0 + rr, gn = n0 + cc;
        if (gk < K && gn < N) v = Bb[(long)gk * ldb + gn];
      }
      ldsB[buf][i] = v;
    }
    __syncthreads();
#endif

    const float* As = &ldsA[buf][0];
    const float* Bs = &ldsB[buf][0];
#pragma unroll
    for (int ks = 0; ks < 2; ++ks) {
      // B fragment (built once, reused for both M-subtiles):
      // lane r = column N, lanes16-31 hold the K+16 half of the substep.
      v16bf fb;
#pragma unroll
      for (int e = 0; e < 8; ++e) {
        int kk = ks * 32 + 16 * hi + 2 * e;
        if (transB) {
          fb[2 * e]     = (__bf16)Bs[(wave * 16 + r) * 64 + kk];
          fb[2 * e + 1] = (__bf16)Bs[(wave * 16 + r) * 64 + kk + 1];
        } else {
          fb[2 * e]     = (__bf16)Bs[kk * 64 + wave * 16 + r];
          fb[2 * e + 1] = (__bf16)Bs[(kk + 1) * 64 + wave * 16 + r];
        }
      }
#pragma unroll
      for (int mi = 0; mi < 2; ++mi) {
        // A fragment: lane r = row, half `hi`
        // selects K ranges {0..15} vs {16..31} (matches B).
        v16bf fa;
#pragma unroll
        for (int e = 0; e < 8; ++e) {
          int kk = ks * 32 + 16 * hi + 2 * e;
          fa[2 * e]     = (__bf16)As[(mi * 16 + r) * 64 + kk];
          fa[2 * e + 1] = (__bf16)As[(mi * 16 + r) * 64 + kk + 1];
        }
        if (mi == 0)
          acc0 = __builtin_amdgcn_wmma_f32_16x16x32_bf16(false, fa, false, fb,
                                                         (short)0, acc0,
                                                         false, false);
        else
          acc1 = __builtin_amdgcn_wmma_f32_16x16x32_bf16(false, fa, false, fb,
                                                         (short)0, acc1,
                                                         false, false);
      }
    }
    buf ^= 1;
  }

  // C/D layout: VGPR e -> row (subtile + e + 8*hi), col n0 + wave*16 + r
  const int col = n0 + wave * 16 + r;
#pragma unroll
  for (int mi = 0; mi < 2; ++mi) {
    v8f acc = (mi == 0) ? acc0 : acc1;
#pragma unroll
    for (int e = 0; e < 8; ++e) {
      int row = m0 + mi * 16 + e + 8 * hi;
      if (row < M && col < N) {
        float val = alpha * acc[e];
        if (bias)  val += bias[col];
        if (accum) val += Cb[(long)row * ldc + col];
        if (relu)  val = val > 0.f ? val : 0.f;
        Cb[(long)row * ldc + col] = val;
      }
    }
  }
}

// ---------------------------------------------------------------------------
// Embedding gather with pad masking.  grid = B*L, block = H
// ---------------------------------------------------------------------------
__global__ __launch_bounds__(H_) void k_embed(const int* __restrict__ ids,
                                              const float* __restrict__ emb,
                                              float* __restrict__ out) {
  int row = blockIdx.x;
  int id = ids[row];
  const float* src = emb + (long)id * H_;
  __builtin_prefetch(src, 0, 0);  // global_prefetch_b8
  float keep = (id != 0) ? 1.f : 0.f;
  out[(long)row * H_ + threadIdx.x] = keep * src[threadIdx.x];
}

// ---------------------------------------------------------------------------
// Row LayerNorm over H=256.  grid = rows, block = 256
// ---------------------------------------------------------------------------
__global__ __launch_bounds__(256) void k_layernorm(const float* __restrict__ x,
                                                   const float* __restrict__ g,
                                                   const float* __restrict__ b,
                                                   float* __restrict__ y) {
  __shared__ float red[256];
  int row = blockIdx.x, t = threadIdx.x;
  float v = x[(long)row * H_ + t];
  red[t] = v;
  __syncthreads();
  for (int s = 128; s > 0; s >>= 1) { if (t < s) red[t] += red[t + s]; __syncthreads(); }
  float mean = red[0] / (float)H_;
  __syncthreads();
  float d = v - mean;
  red[t] = d * d;
  __syncthreads();
  for (int s = 128; s > 0; s >>= 1) { if (t < s) red[t] += red[t + s]; __syncthreads(); }
  float var = red[0] / (float)H_;
  y[(long)row * H_ + t] = d * rsqrtf(var + EPS_) * g[t] + b[t];
}

// ---------------------------------------------------------------------------
// S[h,b,q,k] += Tk[h,b,q,tm[b,q,k]] + Dk[h,b,q,dm[b,q,k]]   (pre-scaled)
// ---------------------------------------------------------------------------
__global__ void k_rel_gather_add(float* __restrict__ S,
                                 const float* __restrict__ Tk,
                                 const float* __restrict__ Dk,
                                 const int* __restrict__ tm,
                                 const int* __restrict__ dm) {
  long idx = (long)blockIdx.x * blockDim.x + threadIdx.x;
  long total = (long)NH_ * B_ * L_ * L_;
  if (idx >= total) return;
  int  k  = (int)(idx % L_);
  long r  = idx / L_;                    // (h*B + b)*L + q
  int  q  = (int)(r % L_);
  long hb = r / L_;
  int  b  = (int)(hb % B_);
  int  h  = (int)(hb / B_);
  long bq = (long)b * L_ + q;
  long base = ((long)h * BL_ + bq) * CPAD_;
  S[idx] += Tk[base + tm[bq * L_ + k]] + Dk[base + dm[bq * L_ + k]];
}

// ---------------------------------------------------------------------------
// Mask (pad | causal) -> NEG, then softmax over k (row of L).  grid = NH*B*L
// ---------------------------------------------------------------------------
__global__ __launch_bounds__(256) void k_softmax(float* __restrict__ S,
                                                 const int* __restrict__ ids) {
  __shared__ float red[256];
  long row = blockIdx.x;                 // (h*B + b)*L + q
  int  q = (int)(row % L_);
  int  b = (int)((row / L_) % B_);
  int  t = threadIdx.x;
  bool valid = t < L_;
  long off = row * L_ + t;
  float s = NEGF;
  if (valid) {
    bool pad = (ids[b * L_ + t] == 0);
    s = (pad || (t > q)) ? NEGF : S[off];
  }
  red[t] = valid ? s : NEGF;
  __syncthreads();
  for (int w = 128; w > 0; w >>= 1) { if (t < w) red[t] = fmaxf(red[t], red[t + w]); __syncthreads(); }
  float mx = red[0];
  __syncthreads();
  float e = valid ? __expf(s - mx) : 0.f;
  red[t] = e;
  __syncthreads();
  for (int w = 128; w > 0; w >>= 1) { if (t < w) red[t] += red[t + w]; __syncthreads(); }
  float inv = 1.f / red[0];
  if (valid) S[off] = e * inv;
}

// ---------------------------------------------------------------------------
// Per (b,q): Wt[h,bq,c] = sum_k A[h,b,q,k]*(tm==c), same for dis. LDS atomics.
// ---------------------------------------------------------------------------
__global__ __launch_bounds__(256) void k_bincount(const float* __restrict__ A,
                                                  const int* __restrict__ tm,
                                                  const int* __restrict__ dm,
                                                  float* __restrict__ Wt,
                                                  float* __restrict__ Wd) {
  __shared__ float binsT[NH_][C_];
  __shared__ float binsD[NH_][C_];
  int bq = blockIdx.x;
  int b = bq / L_, q = bq % L_;
  int t = threadIdx.x;
  for (int i = t; i < NH_ * C_; i += 256) {
    (&binsT[0][0])[i] = 0.f;
    (&binsD[0][0])[i] = 0.f;
  }
  __syncthreads();
  for (int k = t; k < L_; k += 256) {
    int ct = tm[(long)bq * L_ + k];
    int cd = dm[(long)bq * L_ + k];
#pragma unroll
    for (int h = 0; h < NH_; ++h) {
      float a = A[(((long)h * B_ + b) * L_ + q) * L_ + k];
      atomicAdd(&binsT[h][ct], a);   // ds_add_f32
      atomicAdd(&binsD[h][cd], a);
    }
  }
  __syncthreads();
  for (int i = t; i < NH_ * C_; i += 256) {
    int h = i / C_, c = i % C_;
    long o = ((long)h * BL_ + bq) * CPAD_ + c;
    Wt[o] = binsT[h][c];
    Wd[o] = binsD[h][c];
  }
}

// ---------------------------------------------------------------------------
// Elementwise glue
// ---------------------------------------------------------------------------
__global__ void k_add(const float* __restrict__ a, const float* __restrict__ b,
                      float* __restrict__ y, int n) {
  int i = blockIdx.x * blockDim.x + threadIdx.x;
  if (i < n) y[i] = a[i] + b[i];
}
__global__ void k_copy(const float* __restrict__ a, float* __restrict__ y, int n) {
  int i = blockIdx.x * blockDim.x + threadIdx.x;
  if (i < n) y[i] = a[i];
}
__global__ __launch_bounds__(H_) void k_maskmul(float* __restrict__ x,
                                                const int* __restrict__ ids) {
  int row = blockIdx.x;
  float keep = (ids[row] != 0) ? 1.f : 0.f;
  x[(long)row * H_ + threadIdx.x] *= keep;
}

// ---------------------------------------------------------------------------
// Orchestration
// ---------------------------------------------------------------------------
extern "C" void kernel_launch(void* const* d_in, const int* in_sizes, int n_in,
                              void* d_out, int out_size, void* d_ws, size_t ws_size,
                              hipStream_t stream) {
  (void)in_sizes; (void)n_in; (void)out_size; (void)ws_size;
  const int*   log_seqs = (const int*)d_in[0];
  const int*   tm   = (const int*)d_in[1];
  const int*   dm   = (const int*)d_in[2];
  const float* item = (const float*)d_in[3];
  const float* posK = (const float*)d_in[4];
  const float* posV = (const float*)d_in[5];
  const float* timeK = (const float*)d_in[6];
  const float* timeV = (const float*)d_in[7];
  const float* disK = (const float*)d_in[8];
  const float* disV = (const float*)d_in[9];
  const float* ln1g = (const float*)d_in[10];
  const float* ln1b = (const float*)d_in[11];
  const float* Wq = (const float*)d_in[12];
  const float* bq = (const float*)d_in[13];
  const float* Wk = (const float*)d_in[14];
  const float* bk = (const float*)d_in[15];
  const float* Wv = (const float*)d_in[16];
  const float* bv = (const float*)d_in[17];
  const float* ln2g = (const float*)d_in[18];
  const float* ln2b = (const float*)d_in[19];
  const float* W1 = (const float*)d_in[20];
  const float* b1 = (const float*)d_in[21];
  const float* W2 = (const float*)d_in[22];
  const float* b2 = (const float*)d_in[23];
  const float* lnfg = (const float*)d_in[24];
  const float* lnfb = (const float*)d_in[25];

  // Workspace carve-up (~23 MB of fp32)
  float* p = (float*)d_ws;
  float* seqs = p; p += BLH_;
  float* xln  = p; p += BLH_;   // Qn (LN1 out)
  float* x2   = p; p += BLH_;   // LN2 out
  float* Qb   = p; p += BLH_;
  float* Kb   = p; p += BLH_;
  float* Vb   = p; p += BLH_;
  float* O    = p; p += BLH_;
  float* h1   = p; p += BLH_;
  float* S    = p; p += (long)NH_ * B_ * L_ * L_;
  float* Tk   = p; p += (long)NH_ * BL_ * CPAD_;
  float* Dk   = p; p += (long)NH_ * BL_ * CPAD_;
  float* Wt   = p; p += (long)NH_ * BL_ * CPAD_;
  float* Wd   = p; p += (long)NH_ * BL_ * CPAD_;

  auto gemm = [&](const float* A, int lda, long sAh, long sAb,
                  const float* Bm, int ldb, long sBh, long sBb, int transB,
                  const float* bias, float* Cc, int ldc, long sCh, long sCb,
                  int M, int N, int K, float alpha, int accum, int relu,
                  int nz, int nb) {
    dim3 g((M + 31) / 32, (N + 63) / 64, nz);
    hipLaunchKernelGGL(k_gemm, g, dim3(128), 0, stream,
                       A, lda, sAh, sAb, Bm, ldb, sBh, sBb, transB, bias,
                       Cc, ldc, sCh, sCb, M, N, K, alpha, accum, relu, nb);
  };

  const long LH = (long)L_ * H_;       // batch stride in Q/K/V
  const long LL = (long)L_ * L_;
  const long H2 = (long)H_ * H_;

  hipLaunchKernelGGL(k_embed, dim3(BL_), dim3(H_), 0, stream, log_seqs, item, seqs);

  for (int i = 0; i < NB_; ++i) {
    // ---- attention block ----
    hipLaunchKernelGGL(k_layernorm, dim3(BL_), dim3(256), 0, stream,
                       seqs, ln1g + i * H_, ln1b + i * H_, xln);
    gemm(xln,  H_, 0, 0, Wq + (long)i * H2, H_, 0, 0, 1, bq + i * H_,
         Qb, H_, 0, 0, BL_, H_, H_, 1.f, 0, 0, 1, 1);
    gemm(seqs, H_, 0, 0, Wk + (long)i * H2, H_, 0, 0, 1, bk + i * H_,
         Kb, H_, 0, 0, BL_, H_, H_, 1.f, 0, 0, 1, 1);
    gemm(seqs, H_, 0, 0, Wv + (long)i * H2, H_, 0, 0, 1, bv + i * H_,
         Vb, H_, 0, 0, BL_, H_, H_, 1.f, 0, 0, 1, 1);

    // S = (Q K^T)/8  per (h,b);  S += (Q pK^T)/8  (pK shared over b)
    gemm(Qb, H_, 64, LH, Kb,   H_, 64, LH, 1, nullptr,
         S, L_, (long)B_ * LL, LL, L_, L_, HS_, 0.125f, 0, 0, NH_ * B_, B_);
    gemm(Qb, H_, 64, LH, posK, H_, 64, 0,  1, nullptr,
         S, L_, (long)B_ * LL, LL, L_, L_, HS_, 0.125f, 1, 0, NH_ * B_, B_);
    // 257-bin projections: Tk/Dk[h, bq, c] = (Q_head . tab_row_c)/8
    gemm(Qb, H_, 64, 0, timeK, H_, 64, 0, 1, nullptr,
         Tk, CPAD_, (long)BL_ * CPAD_, 0, BL_, C_, HS_, 0.125f, 0, 0, NH_, 1);
    gemm(Qb, H_, 64, 0, disK,  H_, 64, 0, 1, nullptr,
         Dk, CPAD_, (long)BL_ * CPAD_, 0, BL_, C_, HS_, 0.125f, 0, 0, NH_, 1);

    long tot = (long)NH_ * B_ * LL;
    hipLaunchKernelGGL(k_rel_gather_add, dim3((tot + 255) / 256), dim3(256), 0,
                       stream, S, Tk, Dk, tm, dm);
    hipLaunchKernelGGL(k_softmax, dim3(NH_ * BL_), dim3(256), 0, stream, S, log_seqs);
    hipLaunchKernelGGL(k_bincount, dim3(BL_), dim3(256), 0, stream, S, tm, dm, Wt, Wd);

    // O = A V + A pV + Wt timeV + Wd disV
    gemm(S, L_, (long)B_ * LL, LL, Vb,   H_, 64, LH, 0, nullptr,
         O, H_, 64, LH, L_, HS_, L_, 1.f, 0, 0, NH_ * B_, B_);
    gemm(S, L_, (long)B_ * LL, LL, posV, H_, 64, 0,  0, nullptr,
         O, H_, 64, LH, L_, HS_, L_, 1.f, 1, 0, NH_ * B_, B_);
    gemm(Wt, CPAD_, (long)BL_ * CPAD_, 0, timeV, H_, 64, 0, 0, nullptr,
         O, H_, 64, 0, BL_, HS_, C_, 1.f, 1, 0, NH_, 1);
    gemm(Wd, CPAD_, (long)BL_ * CPAD_, 0, disV,  H_, 64, 0, 0, nullptr,
         O, H_, 64, 0, BL_, HS_, C_, 1.f, 1, 0, NH_, 1);

    // seqs = Qn + mha
    hipLaunchKernelGGL(k_add, dim3((BLH_ + 255) / 256), dim3(256), 0, stream,
                       xln, O, seqs, BLH_);

    // ---- FFN ----
    hipLaunchKernelGGL(k_layernorm, dim3(BL_), dim3(256), 0, stream,
                       seqs, ln2g + i * H_, ln2b + i * H_, x2);
    gemm(x2, H_, 0, 0, W1 + (long)i * H2, H_, 0, 0, 1, b1 + i * H_,
         h1, H_, 0, 0, BL_, H_, H_, 1.f, 0, 1, 1, 1);   // relu
    hipLaunchKernelGGL(k_copy, dim3((BLH_ + 255) / 256), dim3(256), 0, stream,
                       x2, seqs, BLH_);                  // seqs = x2
    gemm(h1, H_, 0, 0, W2 + (long)i * H2, H_, 0, 0, 1, b2 + i * H_,
         seqs, H_, 0, 0, BL_, H_, H_, 1.f, 1, 0, 1, 1); // seqs += h1 W2^T + b2
    hipLaunchKernelGGL(k_maskmul, dim3(BL_), dim3(H_), 0, stream, seqs, log_seqs);
  }

  hipLaunchKernelGGL(k_layernorm, dim3(BL_), dim3(256), 0, stream,
                     seqs, lnfg, lnfb, (float*)d_out);
}
